// DeformConv2d_44023414784068
// MI455X (gfx1250) — compile-verified, matrix operand order checked by reading, not run on hardware
//
#include <hip/hip_runtime.h>
#include <hip/hip_bf16.h>

typedef __attribute__((ext_vector_type(16))) _Float16 v16h;
typedef __attribute__((ext_vector_type(8)))  _Float16 v8h;
typedef __attribute__((ext_vector_type(8)))  float    v8f;

#define BB   4
#define CC   64
#define OC   64
#define HH   128
#define WW   128
#define KKT  9      // 3x3 taps
#define KDIM 576    // C * 9
#define NT   4      // 4 n-tiles of 16 -> 64 out channels (main conv)
#define NTO  2      // 2 n-tiles of 16 -> 18 offset channels padded to 32
#define KT   18     // 18 k-tiles of 32 -> 576
#define ASTR 72     // padded halves per A-row in LDS (multiple of 8 -> 16B aligned)
#define OSTR 129    // padded floats per out-row in LDS (odd -> conflict-free)

// LDS plan for main kernel (time-multiplexed):
//   phase 1/2: geom  [0 .. 13824)   + sampA [13824 .. 32256)
//   epilogue : obuf  [0 .. 33024)   (64 rows x 129 floats)
#define SMEM_BYTES 33024
#define SAMPA_OFF  13824

// ---------------------------------------------------------------------------
// Pack conv weight [O=64][C*9=576] f32 -> f16 in the WMMA B-operand layout.
// Tile T = kt*4+nt holds K=kt*32..+31, N=nt*16..+15; per ISA 7.12.2 lane
// l(<16): col N=l, halves 0..7 -> K=0..7, halves 8..15 -> K=16..23; lane l+16:
// same col, K+8.  Each lane's 16 halves are contiguous -> 2x global_load_b128.
// ---------------------------------------------------------------------------
__global__ void pack_w_kernel(const float* __restrict__ weight,
                              _Float16* __restrict__ wp) {
    int idx = blockIdx.x * 256 + threadIdx.x;          // 18*4*32*16 = 36864
    if (idx >= KT * NT * 32 * 16) return;
    int h    = idx & 15;
    int lane = (idx >> 4) & 31;
    int nt   = (idx >> 9) & 3;
    int kt   = idx >> 11;
    int n  = lane & 15;
    int hi = lane >> 4;
    int kl = (h & 7) + ((h >= 8) ? 16 : 0) + hi * 8;   // K within 32-tile
    wp[idx] = (_Float16)weight[(nt * 16 + n) * KDIM + kt * 32 + kl];
}

// Pack offset-conv weight [18][576] f32 -> f16, N padded 18 -> 32 with zeros.
// Tile T = kt*2+nt.
__global__ void pack_ow_kernel(const float* __restrict__ ow,
                               _Float16* __restrict__ wpo) {
    int idx = blockIdx.x * 256 + threadIdx.x;          // 18*2*32*16 = 18432
    if (idx >= KT * NTO * 32 * 16) return;
    int h    = idx & 15;
    int lane = (idx >> 4) & 31;
    int nt   = (idx >> 9) & 1;
    int kt   = idx >> 10;
    int n  = lane & 15;
    int hi = lane >> 4;
    int kl = (h & 7) + ((h >= 8) ? 16 : 0) + hi * 8;
    int o  = nt * 16 + n;
    wpo[idx] = (o < 18) ? (_Float16)ow[o * KDIM + kt * 32 + kl] : (_Float16)0.f;
}

// ---------------------------------------------------------------------------
// Kernel 1: offset conv as WMMA im2col-GEMM.  One workgroup per (b,h):
// 128 pixels x 32 (18 valid) offset channels.  Same pipeline as the main
// kernel but the A-tile fill is a plain zero-padded 3x3 neighborhood copy.
// ---------------------------------------------------------------------------
__global__ __launch_bounds__(256)
void offset_wmma_kernel(const float* __restrict__ x,
                        const _Float16* __restrict__ wpo,
                        const float* __restrict__ ob,
                        float* __restrict__ offsets) {
    __shared__ __align__(16) _Float16 sampA[WW * ASTR];   // 18432 B

    int tid  = threadIdx.x;
    int b    = blockIdx.x / HH;
    int h    = blockIdx.x % HH;
    int lane = tid & 31;
    int wave = tid >> 5;
    int m0   = wave * 16;
    int row  = m0 + (lane & 15);
    int hi   = lane >> 4;

    const float* xb = x + (size_t)b * CC * HH * WW;
    const v16h*  wB = (const v16h*)wpo;   // [kt*2+nt][lane] x 16 halves

    v8f acc[NTO];
    #pragma unroll
    for (int nt = 0; nt < NTO; ++nt) acc[nt] = (v8f){0.f,0.f,0.f,0.f,0.f,0.f,0.f,0.f};

    auto fill_stage = [&](int ks) {
        for (int e = tid; e < WW * 64; e += 256) {
            int px = e & (WW - 1);
            int kj = e >> 7;
            int kg = ks * 64 + kj;
            int c  = kg / 9;
            int kk = kg - c * 9;
            int yy = h - 1 + kk / 3;
            int xx = px - 1 + kk % 3;
            float v = (yy >= 0 && yy < HH && xx >= 0 && xx < WW)
                        ? xb[c * HH * WW + yy * WW + xx] : 0.f;
            sampA[px * ASTR + kj] = (_Float16)v;
        }
    };

    auto do_stage = [&](int ks, v16h* bc, v16h* bn) {
        if (ks < 8) {
            #pragma unroll
            for (int i = 0; i < 4; ++i)     // stage tiles: T = ks*4 + 2*sub + nt
                bn[i] = wB[(((ks + 1) * 4) + i) * 32 + lane];
        }
        fill_stage(ks);
        __syncthreads();

        const _Float16* ap = &sampA[row * ASTR + hi * 8];
        v8h a0lo = *(const v8h*)(ap);
        v8h a0hi = *(const v8h*)(ap + 16);
        v8h a1lo = *(const v8h*)(ap + 32);
        v8h a1hi = *(const v8h*)(ap + 48);
        v16h a0, a1;
        #pragma unroll
        for (int i = 0; i < 8; ++i) {
            a0[i] = a0lo[i]; a0[8 + i] = a0hi[i];
            a1[i] = a1lo[i]; a1[8 + i] = a1hi[i];
        }
        acc[0] = __builtin_amdgcn_wmma_f32_16x16x32_f16(false, a0, false, bc[0], (short)0, acc[0], false, false);
        acc[1] = __builtin_amdgcn_wmma_f32_16x16x32_f16(false, a0, false, bc[1], (short)0, acc[1], false, false);
        acc[0] = __builtin_amdgcn_wmma_f32_16x16x32_f16(false, a1, false, bc[2], (short)0, acc[0], false, false);
        acc[1] = __builtin_amdgcn_wmma_f32_16x16x32_f16(false, a1, false, bc[3], (short)0, acc[1], false, false);
        __syncthreads();
    };

    v16h bA[4], bBuf[4];
    #pragma unroll
    for (int i = 0; i < 4; ++i) bA[i] = wB[i * 32 + lane];

    #pragma unroll 1
    for (int s = 0; s < 8; s += 2) {
        do_stage(s,     bA,   bBuf);
        do_stage(s + 1, bBuf, bA);
    }
    do_stage(8, bA, bBuf);

    // Epilogue: only channels t < 18 are real.
    #pragma unroll
    for (int nt = 0; nt < NTO; ++nt) {
        int t = nt * 16 + (lane & 15);
        if (t < 18) {
            float bv = ob[t];
            size_t base = (((size_t)b * 18 + t) * HH + h) * WW + m0 + hi * 8;
            float4 r0 = make_float4(acc[nt][0] + bv, acc[nt][1] + bv,
                                    acc[nt][2] + bv, acc[nt][3] + bv);
            float4 r1 = make_float4(acc[nt][4] + bv, acc[nt][5] + bv,
                                    acc[nt][6] + bv, acc[nt][7] + bv);
            *(float4*)(offsets + base)     = r0;
            *(float4*)(offsets + base + 4) = r1;
        }
    }
}

// ---------------------------------------------------------------------------
// Kernel 2: fused deform-sample + WMMA GEMM + bias.
// One workgroup per (b, h): 128 pixels x 64 out channels, 8 waves.
// Wave v owns pixels [16v, 16v+16) and all 64 out channels (4x v8f acc).
// K = 576 in 9 stages of 64; B operands double-buffered in registers so the
// fill phase's bilinear VALU work hides all weight-fetch latency.
// ---------------------------------------------------------------------------
struct Geom { float wy, wx; short y0, x0; };

__global__ __launch_bounds__(256)
void deform_wmma_kernel(const float* __restrict__ x,
                        const float* __restrict__ offsets,
                        const _Float16* __restrict__ wp,
                        const float* __restrict__ bias,
                        float* __restrict__ out) {
    __shared__ __align__(16) char smem[SMEM_BYTES];
    Geom*     geom  = (Geom*)smem;
    _Float16* sampA = (_Float16*)(smem + SAMPA_OFF);
    float*    obuf  = (float*)smem;

    int tid  = threadIdx.x;
    int b    = blockIdx.x / HH;
    int h    = blockIdx.x % HH;
    int lane = tid & 31;
    int wave = tid >> 5;

    const float* xb   = x + (size_t)b * CC * HH * WW;
    const float* offs = offsets + (size_t)b * 18 * HH * WW;

    // ---- Phase A: bilinear geometry per (pixel, tap) ----
    for (int e = tid; e < KKT * WW; e += 256) {
        int px = e & (WW - 1);
        int kk = e >> 7;
        float dy = offs[(2 * kk) * HH * WW + h * WW + px];
        float dx = offs[(2 * kk + 1) * HH * WW + h * WW + px];
        float py  = (float)(h - 1 + kk / 3) + dy;
        float pxx = (float)(px - 1 + kk % 3) + dx;
        float y0f = floorf(py);
        float x0f = floorf(pxx);
        Geom g;
        g.wy = py - y0f;
        g.wx = pxx - x0f;
        g.y0 = (short)(int)y0f;
        g.x0 = (short)(int)x0f;
        geom[e] = g;
    }
    __syncthreads();

    v8f acc[NT];
    #pragma unroll
    for (int nt = 0; nt < NT; ++nt) acc[nt] = (v8f){0.f,0.f,0.f,0.f,0.f,0.f,0.f,0.f};

    int m0  = wave * 16;
    int row = m0 + (lane & 15);
    int hi  = lane >> 4;

    const v16h* wB = (const v16h*)wp;   // [kt*4+nt][lane] x 16 halves

    auto fill_stage = [&](int ks) {
        for (int e = tid; e < WW * 64; e += 256) {
            int px = e & (WW - 1);
            int kj = e >> 7;
            int kg = ks * 64 + kj;
            int c  = kg / 9;
            int kk = kg - c * 9;
            Geom g = geom[kk * WW + px];
            int y0 = g.y0, x0 = g.x0;
            const float* xc = xb + c * HH * WW;
            bool yi0 = (y0 >= 0) & (y0 < HH);
            bool yi1 = (y0 + 1 >= 0) & (y0 + 1 < HH);
            bool xi0 = (x0 >= 0) & (x0 < WW);
            bool xi1 = (x0 + 1 >= 0) & (x0 + 1 < WW);
            int yc0 = min(max(y0, 0), HH - 1), yc1 = min(max(y0 + 1, 0), HH - 1);
            int xc0 = min(max(x0, 0), WW - 1), xc1 = min(max(x0 + 1, 0), WW - 1);
            float v00 = (yi0 & xi0) ? xc[yc0 * WW + xc0] : 0.f;
            float v01 = (yi0 & xi1) ? xc[yc0 * WW + xc1] : 0.f;
            float v10 = (yi1 & xi0) ? xc[yc1 * WW + xc0] : 0.f;
            float v11 = (yi1 & xi1) ? xc[yc1 * WW + xc1] : 0.f;
            float wy = g.wy, wx = g.wx;
            float iy = 1.f - wy, ix = 1.f - wx;
            float s = v00 * iy * ix + v01 * iy * wx + v10 * wy * ix + v11 * wy * wx;
            sampA[px * ASTR + kj] = (_Float16)s;
        }
    };

    auto do_stage = [&](int ks, v16h* bc, v16h* bn) {
        if (ks < 8) {
            #pragma unroll
            for (int i = 0; i < 8; ++i)       // next stage tiles (ks+1)*8 + i
                bn[i] = wB[(((ks + 1) * 8) + i) * 32 + lane];
        }
        fill_stage(ks);
        __syncthreads();

        const _Float16* ap = &sampA[row * ASTR + hi * 8];
        v8h a0lo = *(const v8h*)(ap);        // K = hi*8 + 0..7
        v8h a0hi = *(const v8h*)(ap + 16);   // K = 16 + hi*8 + 0..7
        v8h a1lo = *(const v8h*)(ap + 32);
        v8h a1hi = *(const v8h*)(ap + 48);
        v16h a0, a1;
        #pragma unroll
        for (int i = 0; i < 8; ++i) {
            a0[i] = a0lo[i]; a0[8 + i] = a0hi[i];
            a1[i] = a1lo[i]; a1[8 + i] = a1hi[i];
        }
        #pragma unroll
        for (int nt = 0; nt < NT; ++nt)
            acc[nt] = __builtin_amdgcn_wmma_f32_16x16x32_f16(
                false, a0, false, bc[nt], (short)0, acc[nt], false, false);
        #pragma unroll
        for (int nt = 0; nt < NT; ++nt)
            acc[nt] = __builtin_amdgcn_wmma_f32_16x16x32_f16(
                false, a1, false, bc[4 + nt], (short)0, acc[nt], false, false);
        __syncthreads();   // protect sampA before next stage's fill
    };

    v16h bA[8], bBuf[8];
    #pragma unroll
    for (int i = 0; i < 8; ++i) bA[i] = wB[i * 32 + lane];

    #pragma unroll 1
    for (int s = 0; s < 8; s += 2) {
        do_stage(s,     bA,   bBuf);
        do_stage(s + 1, bBuf, bA);
    }
    do_stage(8, bA, bBuf);

    // ---- Epilogue: registers -> LDS transpose -> coalesced global store ----
    #pragma unroll
    for (int nt = 0; nt < NT; ++nt) {
        int o  = nt * 16 + (lane & 15);
        int mb = m0 + hi * 8;
        #pragma unroll
        for (int r = 0; r < 8; ++r)
            obuf[o * OSTR + mb + r] = acc[nt][r];
    }
    __syncthreads();

    for (int j = 0; j < 8; ++j) {
        int g = tid + j * 256;            // float4 group id, 2048 total
        int o = g >> 5;                   // 32 groups of 4 px per channel
        int w4 = (g & 31) * 4;
        float bv = bias[o];
        const float* src = &obuf[o * OSTR + w4];
        float4 r = make_float4(src[0] + bv, src[1] + bv, src[2] + bv, src[3] + bv);
        *(float4*)(out + (((size_t)b * OC + o) * HH + h) * WW + w4) = r;
    }
}

// ---------------------------------------------------------------------------
extern "C" void kernel_launch(void* const* d_in, const int* in_sizes, int n_in,
                              void* d_out, int out_size, void* d_ws, size_t ws_size,
                              hipStream_t stream) {
    const float* x        = (const float*)d_in[0];
    const float* offset_w = (const float*)d_in[1];
    const float* offset_b = (const float*)d_in[2];
    const float* weight   = (const float*)d_in[3];
    const float* bias     = (const float*)d_in[4];
    float* out = (float*)d_out;

    // workspace: [offsets f32: 4*18*128*128][wp f16: 36864][wpo f16: 18432]
    float*    offsets = (float*)d_ws;
    _Float16* wp      = (_Float16*)(offsets + (size_t)BB * 18 * HH * WW);
    _Float16* wpo     = wp + (size_t)KT * NT * 32 * 16;

    pack_w_kernel <<<(KT * NT  * 32 * 16 + 255) / 256, 256, 0, stream>>>(weight, wp);
    pack_ow_kernel<<<(KT * NTO * 32 * 16 + 255) / 256, 256, 0, stream>>>(offset_w, wpo);
    offset_wmma_kernel<<<BB * HH, 256, 0, stream>>>(x, wpo, offset_b, offsets);
    deform_wmma_kernel<<<BB * HH, 256, 0, stream>>>(x, offsets, wp, bias, out);
}
